// Transformer_36644660970322
// MI455X (gfx1250) — matrix operand order repaired and checked by reading, not compiled, not run on hardware
//
#include <hip/hip_runtime.h>
#include <hip/hip_bf16.h>

// ---------------------------------------------------------------------------
// CDNA5 (gfx1250) transformer forward.  All GEMM-like math goes through
// v_wmma_f32_16x16x32_f16 (wave32, 4-wave blocks, f32->f16 LDS staging via
// b128 loads + packed cvt + b64 LDS stores, f32 accumulation).
// GEMM: 64x128 block tile, each wave owns 32x64 (2x4 WMMA tiles).
// Attention is flash-style per (batch, head).
// ---------------------------------------------------------------------------

typedef __attribute__((ext_vector_type(16))) _Float16 v16h;
typedef __attribute__((ext_vector_type(8)))  float    v8f;
typedef __attribute__((ext_vector_type(4)))  float    f32x4;
typedef __attribute__((ext_vector_type(4)))  _Float16 h16x4;

#define DMODEL 512
#define NTOK   2048      // B*S == B*T
#define SEQ    128
#define NHEAD  8
#define HDK    64
#define FFDIM  2048
#define VOCAB  8192

__device__ __forceinline__ v8f wmma16(v16h a, v16h b, v8f c) {
  // (neg_a, A, neg_b, B, c_mod, C, reuse_a, reuse_b)
  return __builtin_amdgcn_wmma_f32_16x16x32_f16(false, a, false, b, (short)0, c,
                                                false, false);
}

// A fragment: 16x32 f16, rows row0..row0+15 of LDS tile (row stride ld halves),
// K-window [k0, k0+32).  Layout per ISA: lane<16 -> Kbase +0, lane>=16 -> +8;
// vgpr v in 0..3 -> K=2v, v in 4..7 -> K=2(v-4)+16.
__device__ __forceinline__ v16h load_a_frag(const _Float16* s, int ld, int row0,
                                            int k0, int lane) {
  const int m  = lane & 15;
  const int hi = (lane >> 4) * 8;
  v16h a;
#pragma unroll
  for (int v = 0; v < 8; ++v) {
#pragma unroll
    for (int h = 0; h < 2; ++h) {
      const int k = 2 * (v & 3) + h + ((v & 4) ? 16 : 0) + hi;
      a[v * 2 + h] = s[(row0 + m) * ld + k0 + k];
    }
  }
  return a;
}

// B fragment: 32x16 f16. column col0+(lane&15); element e -> K row k0+(e&7)+(e&8?16:0)+8*(lane>>4).
__device__ __forceinline__ v16h load_b_frag(const _Float16* s, int ld, int k0,
                                            int col0, int lane) {
  v16h b;
#pragma unroll
  for (int e = 0; e < 16; ++e) b[e] = s[(k0 + (e & 7) + ((e & 8) ? 16 : 0) + 8 * (lane >> 4)) * ld + col0 + (lane & 15)];
  return b;
}

// B fragment where memory holds the transpose: B[k][n] = s[n*ld + k].
__device__ __forceinline__ v16h load_bT_frag(const _Float16* s, int ld, int k0,
                                             int col0, int lane) {
  v16h b;
#pragma unroll
  for (int e = 0; e < 16; ++e) b[e] = s[(col0 + (lane & 15)) * ld + k0 + (e & 7) + ((e & 8) ? 16 : 0) + 8 * (lane >> 4)];
  return b;
}

// f32x4 global load -> f16x4 LDS store (8B aligned by construction).
__device__ __forceinline__ void stage4(const float* __restrict__ g,
                                       _Float16* __restrict__ s) {
  const f32x4 f = *(const f32x4*)g;
  *(h16x4*)s = __builtin_convertvector(f, h16x4);
}

// ---------------------------------------------------------------------------
// GEMM: C[M,N] = A[M,K] @ B[K,N] (+bias[N]) (+res[M,N]) (relu?)
// M multiple of 64, N multiple of 128, K multiple of 32.
// 128 threads / block, 64x128 tile, each wave owns 32x64 (2x4 WMMA tiles).
// ---------------------------------------------------------------------------
#define SA_LD 36    // 64 rows x (32+4) halves -> 72B rows (8B-aligned stores)
#define SB_LD 136   // 32 rows x (128+8) halves -> 272B rows

__global__ __launch_bounds__(128) void gemm_wmma(
    const float* __restrict__ A, const float* __restrict__ B,
    const float* __restrict__ bias, const float* __restrict__ res,
    float* __restrict__ C, int M, int N, int K, int relu) {
  __shared__ _Float16 sA[64 * SA_LD];
  __shared__ _Float16 sB[32 * SB_LD];
  const int tid  = threadIdx.x;
  const int lane = tid & 31;
  const int wave = tid >> 5;
  const int bn = blockIdx.x * 128;
  const int bm = blockIdx.y * 64;
  const int wm = (wave & 1) * 32;
  const int wn = (wave >> 1) * 64;

  v8f acc[2][4] = {};

  for (int k0 = 0; k0 < K; k0 += 32) {
    __syncthreads();
    // A tile: 64x32 f32 = 512 float4, 4 per thread.
#pragma unroll
    for (int it = 0; it < 4; ++it) {
      const int i = tid + it * 128;
      const int r = i >> 3, c = (i & 7) * 4;
      stage4(&A[(size_t)(bm + r) * K + k0 + c], &sA[r * SA_LD + c]);
    }
    // B tile: 32x128 f32 = 1024 float4, 8 per thread.
#pragma unroll
    for (int it = 0; it < 8; ++it) {
      const int i = tid + it * 128;
      const int r = i >> 5, c = (i & 31) * 4;
      stage4(&B[(size_t)(k0 + r) * N + bn + c], &sB[r * SB_LD + c]);
    }
    __syncthreads();
    v16h af0 = load_a_frag(sA, SA_LD, wm, 0, lane);
    v16h af1 = load_a_frag(sA, SA_LD, wm + 16, 0, lane);
#pragma unroll
    for (int nt = 0; nt < 4; ++nt) {
      v16h bf = load_b_frag(sB, SB_LD, 0, wn + nt * 16, lane);
      acc[0][nt] = wmma16(af0, bf, acc[0][nt]);
      acc[1][nt] = wmma16(af1, bf, acc[1][nt]);
    }
  }

#pragma unroll
  for (int mt = 0; mt < 2; ++mt)
#pragma unroll
    for (int nt = 0; nt < 4; ++nt)
#pragma unroll
      for (int v = 0; v < 8; ++v) {
        const int m = bm + wm + mt * 16 + v + 8 * (lane >> 4);
        const int n = bn + wn + nt * 16 + (lane & 15);
        float x = acc[mt][nt][v];
        if (bias) x += bias[n];
        if (res)  x += res[(size_t)m * N + n];
        if (relu) x = fmaxf(x, 0.f);
        C[(size_t)m * N + n] = x;
      }
}

// ---------------------------------------------------------------------------
// Graph attention == dense per-sentence attention with exp(clip(.)) weights.
// One block per (head, batch).  Q/K/V are [NTOK, DMODEL] with head h at
// columns h*64..h*64+63.  Flash-style: stream 32-wide src tiles; accumulate
// O (f32 WMMA accum) and z (row sums) without max-rescaling (fixed clip).
// ---------------------------------------------------------------------------
#define QV_LD 68    // 128 rows x (64+4) halves -> 136B rows (8B-aligned)
#define SS_LD 34

__global__ __launch_bounds__(128) void attn_wmma(
    const float* __restrict__ Q, const float* __restrict__ K,
    const float* __restrict__ V, float* __restrict__ O, int causal) {
  const int h = blockIdx.x, b = blockIdx.y;
  const int tid = threadIdx.x, lane = tid & 31, wave = tid >> 5;

  __shared__ _Float16 sQ[128 * QV_LD];
  __shared__ _Float16 sK[128 * QV_LD];
  __shared__ _Float16 sV[128 * QV_LD];
  __shared__ _Float16 sS[128 * SS_LD];
  __shared__ float    sZ[128];

  const size_t rowbase = (size_t)(b * SEQ) * DMODEL + (size_t)h * HDK;
  // 128x64 f32 per matrix = 2048 float4, 16 per thread.
#pragma unroll
  for (int it = 0; it < 16; ++it) {
    const int i = tid + it * 128;
    const int r = i >> 4, c = (i & 15) * 4;
    const size_t go = rowbase + (size_t)r * DMODEL + c;
    const int so = r * QV_LD + c;
    stage4(&Q[go], &sQ[so]);
    stage4(&K[go], &sK[so]);
    stage4(&V[go], &sV[so]);
  }
  __syncthreads();

  const int wm = wave * 32;        // 4 waves x 32 dst rows
  v8f oacc[2][4] = {};
  float zacc = 0.f;
  const float scale = 0.125f;      // 1/sqrt(64)

  for (int s0 = 0; s0 < 128; s0 += 32) {
    // ---- scores: S = Q @ K^T for dst rows [wm,wm+32), src [s0,s0+32)
    v8f sc[2][2] = {};
#pragma unroll
    for (int k0 = 0; k0 < 64; k0 += 32) {
      v16h aq0 = load_a_frag(sQ, QV_LD, wm, k0, lane);
      v16h aq1 = load_a_frag(sQ, QV_LD, wm + 16, k0, lane);
#pragma unroll
      for (int nt = 0; nt < 2; ++nt) {
        v16h bk = load_bT_frag(sK, QV_LD, k0, s0 + nt * 16, lane);
        sc[0][nt] = wmma16(aq0, bk, sc[0][nt]);
        sc[1][nt] = wmma16(aq1, bk, sc[1][nt]);
      }
    }
    // ---- exp(clip(.)) (+causal mask) -> f16 score tile in LDS
#pragma unroll
    for (int mt = 0; mt < 2; ++mt)
#pragma unroll
      for (int nt = 0; nt < 2; ++nt)
#pragma unroll
        for (int v = 0; v < 8; ++v) {
          const int m = wm + mt * 16 + v + 8 * (lane >> 4);   // dst (0..127)
          const int n = s0 + nt * 16 + (lane & 15);           // src (0..127)
          float x = sc[mt][nt][v] * scale;
          x = expf(fminf(fmaxf(x, -5.f), 5.f));
          if (causal && n > m) x = 0.f;
          sS[m * SS_LD + (n - s0)] = (_Float16)x;
        }
    __syncthreads();
    // ---- z partial (thread t owns dst row t)
    float zp = 0.f;
#pragma unroll
    for (int c = 0; c < 32; ++c) zp += (float)sS[tid * SS_LD + c];
    zacc += zp;
    // ---- O += S_tile @ V_tile
    v16h as0 = load_a_frag(sS, SS_LD, wm, 0, lane);
    v16h as1 = load_a_frag(sS, SS_LD, wm + 16, 0, lane);
#pragma unroll
    for (int vn = 0; vn < 4; ++vn) {
      v16h bv = load_b_frag(sV, QV_LD, s0, vn * 16, lane);
      oacc[0][vn] = wmma16(as0, bv, oacc[0][vn]);
      oacc[1][vn] = wmma16(as1, bv, oacc[1][vn]);
    }
    __syncthreads();
  }

  sZ[tid] = zacc;
  __syncthreads();

#pragma unroll
  for (int mt = 0; mt < 2; ++mt)
#pragma unroll
    for (int vn = 0; vn < 4; ++vn)
#pragma unroll
      for (int v = 0; v < 8; ++v) {
        const int m = wm + mt * 16 + v + 8 * (lane >> 4);
        const int n = vn * 16 + (lane & 15);
        O[rowbase + (size_t)m * DMODEL + n] = oacc[mt][vn][v] / sZ[m];
      }
}

// ---------------------------------------------------------------------------
// LayerNorm: one 128-thread block per row of D=512.
// ---------------------------------------------------------------------------
__global__ __launch_bounds__(128) void ln_kernel(
    const float* __restrict__ x, const float* __restrict__ g,
    const float* __restrict__ b, float* __restrict__ y) {
  const int row = blockIdx.x, t = threadIdx.x;
  const float* xr = x + (size_t)row * DMODEL;
  __shared__ float red[128];
  float v[4];
  float s = 0.f;
#pragma unroll
  for (int i = 0; i < 4; ++i) { v[i] = xr[t + i * 128]; s += v[i]; }
  red[t] = s; __syncthreads();
  for (int o = 64; o > 0; o >>= 1) { if (t < o) red[t] += red[t + o]; __syncthreads(); }
  const float mu = red[0] * (1.f / DMODEL);
  __syncthreads();
  s = 0.f;
#pragma unroll
  for (int i = 0; i < 4; ++i) { const float d = v[i] - mu; s += d * d; }
  red[t] = s; __syncthreads();
  for (int o = 64; o > 0; o >>= 1) { if (t < o) red[t] += red[t + o]; __syncthreads(); }
  const float inv = rsqrtf(red[0] * (1.f / DMODEL) + 1e-5f);
#pragma unroll
  for (int i = 0; i < 4; ++i) {
    const int c = t + i * 128;
    y[(size_t)row * DMODEL + c] = g[c] * (v[i] - mu) * inv + b[c];
  }
}

// ---------------------------------------------------------------------------
// Embedding + sinusoidal positional encoding.
// ---------------------------------------------------------------------------
__global__ __launch_bounds__(256) void embed_kernel(
    const int* __restrict__ tok, const float* __restrict__ emb,
    float* __restrict__ out, int ntokens) {
  const int gid = blockIdx.x * 256 + threadIdx.x;
  if (gid >= ntokens * DMODEL) return;
  const int r = gid >> 9;          // token index
  const int d = gid & 511;
  const int pos = r & (SEQ - 1);
  const float div = expf((float)(d & ~1) * (-9.210340371976184f / DMODEL)); // ln(1e4)
  const float ang = (float)pos * div;
  const float pe  = (d & 1) ? cosf(ang) : sinf(ang);
  out[gid] = emb[(size_t)tok[r] * DMODEL + d] * 22.62741699796952f + pe; // sqrt(512)
}

// ---------------------------------------------------------------------------
// In-place log_softmax over rows of width VOCAB.
// ---------------------------------------------------------------------------
__global__ __launch_bounds__(256) void log_softmax_kernel(float* __restrict__ x) {
  const int row = blockIdx.x, t = threadIdx.x;
  float* xr = x + (size_t)row * VOCAB;
  __shared__ float red[256];
  float mx = -3.4e38f;
  for (int i = t; i < VOCAB; i += 256) mx = fmaxf(mx, xr[i]);
  red[t] = mx; __syncthreads();
  for (int o = 128; o > 0; o >>= 1) { if (t < o) red[t] = fmaxf(red[t], red[t + o]); __syncthreads(); }
  mx = red[0]; __syncthreads();
  float s = 0.f;
  for (int i = t; i < VOCAB; i += 256) s += expf(xr[i] - mx);
  red[t] = s; __syncthreads();
  for (int o = 128; o > 0; o >>= 1) { if (t < o) red[t] += red[t + o]; __syncthreads(); }
  const float lse = mx + logf(red[0]);
  __syncthreads();
  for (int i = t; i < VOCAB; i += 256) xr[i] = xr[i] - lse;
}

// ---------------------------------------------------------------------------
// Host orchestration.
// ---------------------------------------------------------------------------
extern "C" void kernel_launch(void* const* d_in, const int* in_sizes, int n_in,
                              void* d_out, int out_size, void* d_ws, size_t ws_size,
                              hipStream_t stream) {
  (void)in_sizes; (void)n_in; (void)out_size; (void)ws_size;

  const int*   src_tok    = (const int*)d_in[0];
  const int*   tgt_tok    = (const int*)d_in[1];
  // d_in[2..7]: edge lists (structure is exploited analytically, unused)
  const float* src_emb    = (const float*)d_in[8];
  const float* tgt_emb    = (const float*)d_in[9];
  const float* enc_attn_W = (const float*)d_in[10];
  const float* enc_attn_b = (const float*)d_in[11];
  const float* enc_ln_g   = (const float*)d_in[12];
  const float* enc_ln_b   = (const float*)d_in[13];
  const float* enc_ff_W1  = (const float*)d_in[14];
  const float* enc_ff_b1  = (const float*)d_in[15];
  const float* enc_ff_W2  = (const float*)d_in[16];
  const float* enc_ff_b2  = (const float*)d_in[17];
  const float* enc_norm_g = (const float*)d_in[18];
  const float* enc_norm_b = (const float*)d_in[19];
  const float* dec_self_W = (const float*)d_in[20];
  const float* dec_self_b = (const float*)d_in[21];
  const float* dec_cross_W= (const float*)d_in[22];
  const float* dec_cross_b= (const float*)d_in[23];
  const float* dec_ln_g   = (const float*)d_in[24];
  const float* dec_ln_b   = (const float*)d_in[25];
  const float* dec_ff_W1  = (const float*)d_in[26];
  const float* dec_ff_b1  = (const float*)d_in[27];
  const float* dec_ff_W2  = (const float*)d_in[28];
  const float* dec_ff_b2  = (const float*)d_in[29];
  const float* dec_norm_g = (const float*)d_in[30];
  const float* dec_norm_b = (const float*)d_in[31];
  const float* gen_W      = (const float*)d_in[32];
  const float* gen_b      = (const float*)d_in[33];

  const size_t NM = (size_t)NTOK * DMODEL;       // 1,048,576 floats
  float* ws  = (float*)d_ws;
  float* xe  = ws;
  float* xd  = ws + 1 * NM;
  float* nx  = ws + 2 * NM;
  float* nkv = ws + 3 * NM;
  float* q   = ws + 4 * NM;
  float* k   = ws + 5 * NM;
  float* v   = ws + 6 * NM;
  float* att = ws + 7 * NM;
  float* h1  = ws + 8 * NM;                      // NTOK x FFDIM (4*NM)
  float* out = (float*)d_out;                    // NTOK x VOCAB logits

  auto gemm = [&](const float* A, const float* B, const float* bias,
                  const float* res, float* C, int M, int N, int K, int relu) {
    dim3 grid(N / 128, M / 64);
    gemm_wmma<<<grid, 128, 0, stream>>>(A, B, bias, res, C, M, N, K, relu);
  };
  auto lnorm = [&](const float* x, const float* g, const float* b, float* y) {
    ln_kernel<<<NTOK, 128, 0, stream>>>(x, g, b, y);
  };
  auto attn = [&](const float* Q, const float* K_, const float* V_, float* O,
                  int causal) {
    attn_wmma<<<dim3(NHEAD, 16), 128, 0, stream>>>(Q, K_, V_, O, causal);
  };

  const size_t WSZ = (size_t)DMODEL * DMODEL;    // 262144

  // ---- embeddings + PE ----
  embed_kernel<<<(NTOK * DMODEL) / 256, 256, 0, stream>>>(src_tok, src_emb, xe, NTOK);
  embed_kernel<<<(NTOK * DMODEL) / 256, 256, 0, stream>>>(tgt_tok, tgt_emb, xd, NTOK);

  // ---- encoder ----
  for (int i = 0; i < 2; ++i) {
    const float* W = enc_attn_W + (size_t)i * 4 * WSZ;
    const float* Bv = enc_attn_b + (size_t)i * 4 * DMODEL;
    lnorm(xe, enc_ln_g + (i * 2 + 0) * DMODEL, enc_ln_b + (i * 2 + 0) * DMODEL, nx);
    gemm(nx, W + 0 * WSZ, Bv + 0 * DMODEL, nullptr, q, NTOK, DMODEL, DMODEL, 0);
    gemm(nx, W + 1 * WSZ, Bv + 1 * DMODEL, nullptr, k, NTOK, DMODEL, DMODEL, 0);
    gemm(nx, W + 2 * WSZ, Bv + 2 * DMODEL, nullptr, v, NTOK, DMODEL, DMODEL, 0);
    attn(q, k, v, att, 0);
    gemm(att, W + 3 * WSZ, Bv + 3 * DMODEL, xe, xe, NTOK, DMODEL, DMODEL, 0);
    lnorm(xe, enc_ln_g + (i * 2 + 1) * DMODEL, enc_ln_b + (i * 2 + 1) * DMODEL, nx);
    gemm(nx, enc_ff_W1 + (size_t)i * DMODEL * FFDIM, enc_ff_b1 + (size_t)i * FFDIM,
         nullptr, h1, NTOK, FFDIM, DMODEL, 1);
    gemm(h1, enc_ff_W2 + (size_t)i * FFDIM * DMODEL, enc_ff_b2 + (size_t)i * DMODEL,
         xe, xe, NTOK, DMODEL, FFDIM, 0);
    if (i == 1) lnorm(xe, enc_norm_g, enc_norm_b, xe);
  }

  // ---- decoder ----
  for (int i = 0; i < 2; ++i) {
    const float* sW = dec_self_W + (size_t)i * 4 * WSZ;
    const float* sB = dec_self_b + (size_t)i * 4 * DMODEL;
    const float* cW = dec_cross_W + (size_t)i * 4 * WSZ;
    const float* cB = dec_cross_b + (size_t)i * 4 * DMODEL;
    // self-attention (causal)
    lnorm(xd, dec_ln_g + (i * 3 + 0) * DMODEL, dec_ln_b + (i * 3 + 0) * DMODEL, nx);
    gemm(nx, sW + 0 * WSZ, sB + 0 * DMODEL, nullptr, q, NTOK, DMODEL, DMODEL, 0);
    gemm(nx, sW + 1 * WSZ, sB + 1 * DMODEL, nullptr, k, NTOK, DMODEL, DMODEL, 0);
    gemm(nx, sW + 2 * WSZ, sB + 2 * DMODEL, nullptr, v, NTOK, DMODEL, DMODEL, 0);
    attn(q, k, v, att, 1);
    gemm(att, sW + 3 * WSZ, sB + 3 * DMODEL, xd, xd, NTOK, DMODEL, DMODEL, 0);
    // cross-attention
    lnorm(xd, dec_ln_g + (i * 3 + 1) * DMODEL, dec_ln_b + (i * 3 + 1) * DMODEL, nx);
    lnorm(xe, dec_ln_g + (i * 3 + 1) * DMODEL, dec_ln_b + (i * 3 + 1) * DMODEL, nkv);
    gemm(nx,  cW + 0 * WSZ, cB + 0 * DMODEL, nullptr, q, NTOK, DMODEL, DMODEL, 0);
    gemm(nkv, cW + 1 * WSZ, cB + 1 * DMODEL, nullptr, k, NTOK, DMODEL, DMODEL, 0);
    gemm(nkv, cW + 2 * WSZ, cB + 2 * DMODEL, nullptr, v, NTOK, DMODEL, DMODEL, 0);
    attn(q, k, v, att, 0);
    gemm(att, cW + 3 * WSZ, cB + 3 * DMODEL, xd, xd, NTOK, DMODEL, DMODEL, 0);
    // FFN
    lnorm(xd, dec_ln_g + (i * 3 + 2) * DMODEL, dec_ln_b + (i * 3 + 2) * DMODEL, nx);
    gemm(nx, dec_ff_W1 + (size_t)i * DMODEL * FFDIM, dec_ff_b1 + (size_t)i * FFDIM,
         nullptr, h1, NTOK, FFDIM, DMODEL, 1);
    gemm(h1, dec_ff_W2 + (size_t)i * FFDIM * DMODEL, dec_ff_b2 + (size_t)i * DMODEL,
         xd, xd, NTOK, DMODEL, FFDIM, 0);
    if (i == 1) lnorm(xd, dec_norm_g, dec_norm_b, xd);
  }

  // ---- generator + log_softmax ----
  gemm(xd, gen_W, gen_b, nullptr, out, NTOK, VOCAB, DMODEL, 0);
  log_softmax_kernel<<<NTOK, 256, 0, stream>>>(out);
}
